// LuongAttnDecoderRNN_10634339025098
// MI455X (gfx1250) — matrix operand order repaired and checked by reading, not compiled, not run on hardware
//
#include <hip/hip_runtime.h>

// ---------------------------------------------------------------------------
// Types for CDNA5 WMMA (wave32, gfx1250)
// ---------------------------------------------------------------------------
typedef __attribute__((ext_vector_type(16))) __bf16          v16bf;
typedef __attribute__((ext_vector_type(8)))  float           v8f;
typedef __attribute__((ext_vector_type(8)))  unsigned short  u16x8;

union Frag16 { u16x8 h[2]; v16bf v; };

__device__ __forceinline__ unsigned short f2bf(float f) {
  unsigned int u = __float_as_uint(f);
  unsigned int lsb = (u >> 16) & 1u;
  u += 0x7FFFu + lsb;                 // round-to-nearest-even
  return (unsigned short)(u >> 16);
}

__device__ __forceinline__ float sigm(float x) {
  return 1.0f / (1.0f + __expf(-x));
}

// ---------------------------------------------------------------------------
// Async global->LDS staging (CDNA5): 16B per lane, tracked by ASYNCcnt.
// Inline asm keeps this portable across ROCm/amdgpu-toolchain builtin skews.
// ---------------------------------------------------------------------------
#define USE_ASYNC_LDS 1

__device__ __forceinline__ void async_g2l_b128(const void* g, void* l) {
#if USE_ASYNC_LDS
  unsigned           lofs = (unsigned)(size_t)l;        // LDS byte address
  unsigned long long ga   = (unsigned long long)(size_t)g;
  asm volatile("global_load_async_to_lds_b128 %0, %1, off"
               :: "v"(lofs), "v"(ga) : "memory");
#else
  *(uint4*)l = *(const uint4*)g;
#endif
}

template <int N>
__device__ __forceinline__ void wait_async() {
#if USE_ASYNC_LDS
#if defined(__has_builtin)
#if __has_builtin(__builtin_amdgcn_s_wait_asynccnt)
  __builtin_amdgcn_s_wait_asynccnt(N);
#else
  asm volatile("s_wait_asynccnt %0" :: "i"(N) : "memory");
#endif
#else
  asm volatile("s_wait_asynccnt %0" :: "i"(N) : "memory");
#endif
#endif
}

// ---------------------------------------------------------------------------
// Generic WMMA GEMM on bf16 operands:
//   C[M,N] = act( A[M,K] @ Bw[N,K]^T + bias[N] ),  C stored f32 or bf16.
// A may be split into 3 row-major parts along K (kChunk wide each) so the
// concat GEMM reads [rnn | u_ctx | b_ctx] without materializing the concat.
// Block tile 64x128, K step 64, 256 threads = 8 waves, 8 wmma per wave/iter.
// Double-buffered LDS: tile i+1 is in flight (async b128 copies, ASYNCcnt<=6)
// while tile i feeds the WMMA pipe.
// ---------------------------------------------------------------------------
#define BM 64
#define BN 128
#define BK 64
#define LDSP 72   // padded LDS row stride in halfwords (144B, keeps 16B align)

__device__ __forceinline__ void stage_tile(
    const unsigned short* __restrict__ Ap, int kChunk, int kl,
    const unsigned short* __restrict__ Bw, int K, int kk,
    int m0, int n0, unsigned short* AsB, unsigned short* BsB, int tid)
{
  // A tile (64x64 bf16): 512 16B-groups, 2 per thread
#pragma unroll
  for (int i = 0; i < 2; ++i) {
    int idx = tid + i * 256;
    int r = idx >> 3, c8 = (idx & 7) * 8;
    async_g2l_b128(Ap + (size_t)(m0 + r) * kChunk + kl + c8, &AsB[r * LDSP + c8]);
  }
  // B tile (128x64 bf16): 1024 16B-groups, 4 per thread
#pragma unroll
  for (int i = 0; i < 4; ++i) {
    int idx = tid + i * 256;
    int r = idx >> 3, c8 = (idx & 7) * 8;
    async_g2l_b128(Bw + (size_t)(n0 + r) * K + kk + c8, &BsB[r * LDSP + c8]);
  }
}

__global__ __launch_bounds__(256) void gemm_bf16_wmma(
    const unsigned short* __restrict__ A0, const unsigned short* __restrict__ A1,
    const unsigned short* __restrict__ A2, int kChunk,
    const unsigned short* __restrict__ Bw, const float* __restrict__ bias,
    float* __restrict__ Cf, unsigned short* __restrict__ Cb,
    int N, int K, int act)
{
  __shared__ unsigned short As[2][BM * LDSP];
  __shared__ unsigned short Bs[2][BN * LDSP];

  const int tid  = threadIdx.x;
  const int lane = tid & 31;
  const int wave = tid >> 5;
  const int m0   = blockIdx.y * BM;
  const int n0   = blockIdx.x * BN;
  const int mw   = (wave >> 1) * 16;   // 4 row strips of 16
  const int nw   = (wave & 1) * 64;    // 2 column halves of 64
  const int lr   = lane & 15;
  const int hi   = lane >> 4;

  v8f acc[4];
#pragma unroll
  for (int i = 0; i < 4; ++i)
#pragma unroll
    for (int j = 0; j < 8; ++j) acc[i][j] = 0.0f;

  const int nIter = K / BK;

  // Prologue: stage tile 0 into buffer 0 (kk=0 is always in part 0)
  stage_tile(A0, kChunk, 0, Bw, K, 0, m0, n0, As[0], Bs[0], tid);

  for (int it = 0; it < nIter; ++it) {
    const int cur = it & 1;

    if (it + 1 < nIter) {
      // Prefetch tile i+1 into the other buffer (released by last barrier),
      // then wait for the *older* 6 copies only (in-order completion).
      const int kk2   = (it + 1) * BK;
      const int part2 = kk2 / kChunk;
      const unsigned short* __restrict__ Ap2 =
          (part2 == 0) ? A0 : ((part2 == 1) ? A1 : A2);
      stage_tile(Ap2, kChunk, kk2 - part2 * kChunk, Bw, K, kk2,
                 m0, n0, As[cur ^ 1], Bs[cur ^ 1], tid);
      wait_async<6>();
    } else {
      wait_async<0>();
    }
    __syncthreads();

#pragma unroll
    for (int ks = 0; ks < BK; ks += 32) {
      // A fragment 16x32: lanes 0-15 hold K 0..7 & 16..23 of the chunk,
      // lanes 16-31 hold 8..15 & 24..31
      Frag16 fa;
      fa.h[0] = *reinterpret_cast<const u16x8*>(&As[cur][(mw + lr) * LDSP + ks + hi * 8]);
      fa.h[1] = *reinterpret_cast<const u16x8*>(&As[cur][(mw + lr) * LDSP + ks + 16 + hi * 8]);

#pragma unroll
      for (int nf = 0; nf < 4; ++nf) {
        // B fragment 32x16: lane holds column n, K 0..7 & 16..23 (lanes 0-15), 8..15 & 24..31 (lanes 16-31)
        Frag16 fb;
        const int nb = nw + nf * 16 + lr;
        fb.h[0] = *reinterpret_cast<const u16x8*>(&Bs[cur][nb * LDSP + ks + hi * 8]);
        fb.h[1] = *reinterpret_cast<const u16x8*>(&Bs[cur][nb * LDSP + ks + 16 + hi * 8]);

        acc[nf] = __builtin_amdgcn_wmma_f32_16x16x32_bf16(
            false, fa.v, false, fb.v, (short)0, acc[nf], false, false);
      }
    }
    __syncthreads();   // release buffer `cur` for the prefetch 2 iters ahead
  }

  // Epilogue: D VGPR r holds row m0+mw+hi*8+r, column n0+nw+nf*16+(lane&15)
#pragma unroll
  for (int nf = 0; nf < 4; ++nf) {
    const int n = n0 + nw + nf * 16 + lr;
    const float bv = bias ? bias[n] : 0.0f;
#pragma unroll
    for (int r = 0; r < 8; ++r) {
      const int m = m0 + mw + hi * 8 + r;
      float v = acc[nf][r] + bv;
      if (act) v = tanhf(v);
      if (Cb) Cb[(size_t)m * N + n] = f2bf(v);
      else    Cf[(size_t)m * N + n] = v;
    }
  }
}

// ---------------------------------------------------------------------------
// Packed f32 -> bf16 conversion: 8 elements per thread, b128 in / b128 out
// ---------------------------------------------------------------------------
__global__ __launch_bounds__(256) void cvt_bf16(
    const float* __restrict__ src, unsigned short* __restrict__ dst, int n8)
{
  const int g = blockIdx.x * 256 + threadIdx.x;
  if (g >= n8) return;
  const size_t base = (size_t)g * 8;
  const float4 a = *(const float4*)(src + base);
  const float4 b = *(const float4*)(src + base + 4);
  union { unsigned short s[8]; uint4 u; } o;
  o.s[0] = f2bf(a.x); o.s[1] = f2bf(a.y); o.s[2] = f2bf(a.z); o.s[3] = f2bf(a.w);
  o.s[4] = f2bf(b.x); o.s[5] = f2bf(b.y); o.s[6] = f2bf(b.z); o.s[7] = f2bf(b.w);
  *(uint4*)(dst + base) = o.u;
}

// ---------------------------------------------------------------------------
// Embedding gather, emitting bf16 directly: X[n*64+b, :] = bf16(emb[seq, :])
// ---------------------------------------------------------------------------
__global__ __launch_bounds__(256) void gather_emb_bf16(
    const int* __restrict__ seq, const float* __restrict__ emb,
    unsigned short* __restrict__ X)
{
  const int g = blockIdx.x * 256 + threadIdx.x;    // 2048*1024/8 groups
  const int row = g >> 7;
  const int c8  = (g & 127) * 8;
  const float* src = emb + (size_t)seq[row] * 1024 + c8;
  const float4 a = *(const float4*)src;
  const float4 b = *(const float4*)(src + 4);
  union { unsigned short s[8]; uint4 u; } o;
  o.s[0] = f2bf(a.x); o.s[1] = f2bf(a.y); o.s[2] = f2bf(a.z); o.s[3] = f2bf(a.w);
  o.s[4] = f2bf(b.x); o.s[5] = f2bf(b.y); o.s[6] = f2bf(b.z); o.s[7] = f2bf(b.w);
  *(uint4*)(X + (size_t)row * 1024 + c8) = o.u;
}

// ---------------------------------------------------------------------------
// GRU gates for one step; writes h and rnn_out in both f32 and bf16.
// ---------------------------------------------------------------------------
__global__ __launch_bounds__(256) void gru_gates(
    const float* __restrict__ gi, const float* __restrict__ gh,
    float* __restrict__ h, unsigned short* __restrict__ hbf,
    float* __restrict__ rnnf, unsigned short* __restrict__ rnnbf, int s)
{
  const int idx = blockIdx.x * 256 + threadIdx.x;   // 64*1024
  const int row = idx >> 10;                        // b
  const int col = idx & 1023;
  const float* gir = gi + (size_t)(s * 64 + row) * 3072;
  const float* ghr = gh + (size_t)row * 3072;
  const float ir = gir[col], iz = gir[1024 + col], inn = gir[2048 + col];
  const float hr = ghr[col], hz = ghr[1024 + col], hn = ghr[2048 + col];
  const float r  = sigm(ir + hr);
  const float z  = sigm(iz + hz);
  const float nn = tanhf(inn + r * hn);
  const float hnew = (1.0f - z) * nn + z * h[idx];
  h[idx]   = hnew;
  hbf[idx] = f2bf(hnew);
  rnnf[(size_t)s * 65536 + idx]  = hnew;
  rnnbf[(size_t)s * 65536 + idx] = f2bf(hnew);
}

// ---------------------------------------------------------------------------
// Attention projections: u[j] = sum_i attn_W[i, j] * v[i] for j in [0,2048)
// (covers both W1^T v and W2^T v since attn_W is (1024, 2048)); u[2048]=b.v
// ---------------------------------------------------------------------------
__global__ __launch_bounds__(256) void attn_uv(
    const float* __restrict__ W, const float* __restrict__ b,
    const float* __restrict__ v, float* __restrict__ u)
{
  const int j = blockIdx.x * 256 + threadIdx.x;
  if (j < 2048) {
    float s = 0.0f;
    for (int i = 0; i < 1024; ++i) s += W[(size_t)i * 2048 + j] * v[i];
    u[j] = s;
  } else if (j == 2048) {
    float s = 0.0f;
    for (int i = 0; i < 1024; ++i) s += b[i] * v[i];
    u[2048] = s;
  }
}

// ---------------------------------------------------------------------------
// Row-vector dot: out[row] = dot(Arows[row, 0:1024], u) (+ *addp). One wave
// per row, lane-strided accumulate + shuffle reduction (wave32).
// ---------------------------------------------------------------------------
__global__ __launch_bounds__(256) void rowdot(
    const float* __restrict__ Arows, const float* __restrict__ u,
    float* __restrict__ out, int nrows, const float* __restrict__ addp)
{
  const int row  = blockIdx.x * 8 + (threadIdx.x >> 5);
  const int lane = threadIdx.x & 31;
  if (row >= nrows) return;
  const float* p = Arows + (size_t)row * 1024;
  float s = 0.0f;
  for (int k = lane; k < 1024; k += 32) s += p[k] * u[k];
#pragma unroll
  for (int off = 16; off >= 1; off >>= 1) s += __shfl_xor(s, off, 32);
  if (lane == 0) {
    float add = addp ? addp[0] : 0.0f;
    out[row] = s + add;
  }
}

// ---------------------------------------------------------------------------
// Softmax over T=400:  w[b,n,t] = softmax_t( a[n,b] + c[t,b] )
// One block per (b,n); writes directly into d_out region.
// ---------------------------------------------------------------------------
__global__ __launch_bounds__(128) void attn_softmax(
    const float* __restrict__ a, const float* __restrict__ c,
    float* __restrict__ w)
{
  const int bn = blockIdx.x;        // b*32 + n
  const int b  = bn >> 5;
  const int n  = bn & 31;
  const int tid = threadIdx.x;
  __shared__ float red[128];

  const float av = a[n * 64 + b];
  float e[4];
  float mx = -3.0e38f;
  int cnt = 0;
  for (int t = tid; t < 400; t += 128) {
    float ev = av + c[t * 64 + b];
    e[cnt++] = ev;
    mx = fmaxf(mx, ev);
  }
  red[tid] = mx;
  __syncthreads();
  for (int s = 64; s >= 1; s >>= 1) {
    if (tid < s) red[tid] = fmaxf(red[tid], red[tid + s]);
    __syncthreads();
  }
  mx = red[0];
  __syncthreads();

  float sum = 0.0f;
  cnt = 0;
  for (int t = tid; t < 400; t += 128) {
    float ex = __expf(e[cnt] - mx);
    e[cnt++] = ex;
    sum += ex;
  }
  red[tid] = sum;
  __syncthreads();
  for (int s = 64; s >= 1; s >>= 1) {
    if (tid < s) red[tid] += red[tid + s];
    __syncthreads();
  }
  const float inv = 1.0f / red[0];

  cnt = 0;
  for (int t = tid; t < 400; t += 128)
    w[(size_t)bn * 400 + t] = e[cnt++] * inv;
}

// ---------------------------------------------------------------------------
// Context: ctx[n,b,:] = bf16( sum_t w[b,n,t] * enc[t,b,:] )
// Block = (b, group of 8 n); 256 threads each own 4 h-columns; w rows staged
// in LDS so the encoder slice is streamed once per 8 outputs.
// ---------------------------------------------------------------------------
__global__ __launch_bounds__(256) void attn_ctx(
    const float* __restrict__ w, const float* __restrict__ enc,
    unsigned short* __restrict__ ctx)
{
  const int b  = blockIdx.x >> 2;
  const int n0 = (blockIdx.x & 3) * 8;
  const int tid = threadIdx.x;
  __shared__ float wsm[8 * 400];

  for (int idx = tid; idx < 3200; idx += 256) {
    int j = idx / 400;
    int t = idx - j * 400;
    wsm[idx] = w[(size_t)(b * 32 + n0 + j) * 400 + t];
  }
  __syncthreads();

  const int h0 = tid * 4;
  float acc[8][4];
#pragma unroll
  for (int j = 0; j < 8; ++j)
#pragma unroll
    for (int k = 0; k < 4; ++k) acc[j][k] = 0.0f;

  for (int t = 0; t < 400; ++t) {
    const float4 ev =
        *reinterpret_cast<const float4*>(&enc[((size_t)t * 64 + b) * 1024 + h0]);
#pragma unroll
    for (int j = 0; j < 8; ++j) {
      const float wj = wsm[j * 400 + t];
      acc[j][0] += wj * ev.x;
      acc[j][1] += wj * ev.y;
      acc[j][2] += wj * ev.z;
      acc[j][3] += wj * ev.w;
    }
  }

#pragma unroll
  for (int j = 0; j < 8; ++j) {
    union { unsigned short s[4]; uint2 u; } o;
    o.s[0] = f2bf(acc[j][0]); o.s[1] = f2bf(acc[j][1]);
    o.s[2] = f2bf(acc[j][2]); o.s[3] = f2bf(acc[j][3]);
    *(uint2*)&ctx[((size_t)(n0 + j) * 64 + b) * 1024 + h0] = o.u;
  }
}

// ---------------------------------------------------------------------------
// Host-side orchestration
// ---------------------------------------------------------------------------
extern "C" void kernel_launch(void* const* d_in, const int* in_sizes, int n_in,
                              void* d_out, int out_size, void* d_ws, size_t ws_size,
                              hipStream_t stream) {
  (void)in_sizes; (void)n_in; (void)out_size; (void)ws_size;

  const int*   seq      = (const int*)  d_in[0];   // (32,64)
  const float* last_h   = (const float*)d_in[1];   // (1,64,1024)
  const float* encU     = (const float*)d_in[2];   // (400,64,1024)
  const float* encB     = (const float*)d_in[3];   // (400,64,1024)
  const float* emb      = (const float*)d_in[4];   // (32000,1024)
  const float* W_ih     = (const float*)d_in[5];   // (3072,1024)
  const float* W_hh     = (const float*)d_in[6];   // (3072,1024)
  const float* b_ih     = (const float*)d_in[7];
  const float* b_hh     = (const float*)d_in[8];
  const float* attn_W   = (const float*)d_in[9];   // (1024,2048)
  const float* attn_b   = (const float*)d_in[10];
  const float* attn_v   = (const float*)d_in[11];
  const float* concat_W = (const float*)d_in[12];  // (1024,3072)
  const float* concat_b = (const float*)d_in[13];
  const float* out_W    = (const float*)d_in[14];  // (32000,1024)
  const float* out_b    = (const float*)d_in[15];

  float* out = (float*)d_out;
  const size_t OUT_HIDDEN = 65536000;                // after 32*64*32000
  const size_t OUT_UW     = 65601536;                // + 64*1024
  const size_t OUT_BW     = 66420736;                // + 64*32*400

  // Workspace carve (byte offsets, 16B aligned throughout)
  char* base = (char*)d_ws;
  size_t off = 0;
  auto allocF = [&](size_t n) { float* p = (float*)(base + off); off += n * 4; return p; };
  auto allocU = [&](size_t n) { unsigned short* p = (unsigned short*)(base + off); off += n * 2; return p; };

  unsigned short* Xbf   = allocU((size_t)2048 * 1024);  // bf16 embeddings
  float*          gi    = allocF((size_t)2048 * 3072);  // x@W_ih^T + b_ih
  float*          gh    = allocF((size_t)64 * 3072);    // per-step h@W_hh^T + b_hh
  float*          hbuf  = allocF((size_t)64 * 1024);    // hidden state f32
  unsigned short* hbf   = allocU((size_t)64 * 1024);    // hidden state bf16
  float*          rnnf  = allocF((size_t)2048 * 1024);  // rnn_out f32
  unsigned short* rnnbf = allocU((size_t)2048 * 1024);  // rnn_out bf16
  float*          u     = allocF(2112);                 // u1 | u2 | bias
  float*          av    = allocF(2048);                 // a (N,B)
  float*          cu    = allocF(25600);                // c encoder U (T,B)
  float*          cb    = allocF(25600);                // c encoder B (T,B)
  unsigned short* uctx  = allocU((size_t)2048 * 1024);  // bf16 context U
  unsigned short* bctx  = allocU((size_t)2048 * 1024);  // bf16 context B
  unsigned short* h2bf  = allocU((size_t)2048 * 1024);  // tanh(concat proj), bf16
  unsigned short* WihB  = allocU((size_t)3072 * 1024);
  unsigned short* WhhB  = allocU((size_t)3072 * 1024);
  unsigned short* cWB   = allocU((size_t)1024 * 3072);
  unsigned short* oWB   = allocU((size_t)32000 * 1024);

  // 0) Weight / state conversions to bf16 (packed b128 in/out)
  cvt_bf16<<<(3072 * 1024 / 8) / 256, 256, 0, stream>>>(W_ih, WihB, 3072 * 1024 / 8);
  cvt_bf16<<<(3072 * 1024 / 8) / 256, 256, 0, stream>>>(W_hh, WhhB, 3072 * 1024 / 8);
  cvt_bf16<<<(3072 * 1024 / 8) / 256, 256, 0, stream>>>(concat_W, cWB, 3072 * 1024 / 8);
  cvt_bf16<<<(32000 * 1024 / 8) / 256, 256, 0, stream>>>(out_W, oWB, 32000 * 1024 / 8);
  cvt_bf16<<<(64 * 1024 / 8) / 256, 256, 0, stream>>>(last_h, hbf, 64 * 1024 / 8);

  // 1) Embedding gather (bf16 out)
  gather_emb_bf16<<<(2048 * 1024 / 8) / 256, 256, 0, stream>>>(seq, emb, Xbf);

  // 2) gi = X @ W_ih^T + b_ih for all 32 steps at once  (2048x3072, K=1024)
  gemm_bf16_wmma<<<dim3(3072 / BN, 2048 / BM), 256, 0, stream>>>(
      Xbf, Xbf, Xbf, 1024, WihB, b_ih, gi, nullptr, 3072, 1024, 0);

  // 3) h0 = last_hidden (f32 side)
  hipMemcpyAsync(hbuf, last_h, (size_t)64 * 1024 * sizeof(float),
                 hipMemcpyDeviceToDevice, stream);

  // 4) GRU recurrence: 32 sequential (gh GEMM + gates) pairs
  for (int s = 0; s < 32; ++s) {
    gemm_bf16_wmma<<<dim3(3072 / BN, 1), 256, 0, stream>>>(
        hbf, hbf, hbf, 1024, WhhB, b_hh, gh, nullptr, 3072, 1024, 0);
    gru_gates<<<(64 * 1024) / 256, 256, 0, stream>>>(gi, gh, hbuf, hbf, rnnf, rnnbf, s);
  }

  // 5) hidden output
  hipMemcpyAsync(out + OUT_HIDDEN, hbuf, (size_t)64 * 1024 * sizeof(float),
                 hipMemcpyDeviceToDevice, stream);

  // 6) attention projections (shared by both encoders)
  attn_uv<<<(2049 + 255) / 256, 256, 0, stream>>>(attn_W, attn_b, attn_v, u);

  // 7) a = rnn_out @ u1 ; c = enc @ u2 + bias
  rowdot<<<2048 / 8, 256, 0, stream>>>(rnnf, u, av, 2048, nullptr);
  rowdot<<<25600 / 8, 256, 0, stream>>>(encU, u + 1024, cu, 25600, u + 2048);
  rowdot<<<25600 / 8, 256, 0, stream>>>(encB, u + 1024, cb, 25600, u + 2048);

  // 8) softmax -> uw / bw written directly into d_out
  attn_softmax<<<2048, 128, 0, stream>>>(av, cu, out + OUT_UW);
  attn_softmax<<<2048, 128, 0, stream>>>(av, cb, out + OUT_BW);

  // 9) contexts (bf16 out)
  attn_ctx<<<64 * 4, 256, 0, stream>>>(out + OUT_UW, encU, uctx);
  attn_ctx<<<64 * 4, 256, 0, stream>>>(out + OUT_BW, encB, bctx);

  // 10) h = tanh([rnn | uctx | bctx] @ concat_W^T + concat_b), bf16 out
  gemm_bf16_wmma<<<dim3(1024 / BN, 2048 / BM), 256, 0, stream>>>(
      rnnbf, uctx, bctx, 1024, cWB, concat_b, nullptr, h2bf, 1024, 3072, 1);

  // 11) logits = h @ out_W^T + out_b  (dominant GEMM, 134 GFLOP) -> f32 d_out
  gemm_bf16_wmma<<<dim3(32000 / BN, 2048 / BM), 256, 0, stream>>>(
      h2bf, h2bf, h2bf, 1024, oWB, out_b, out, nullptr, 32000, 1024, 0);
}
